// GNNDependency_65025804861526
// MI455X (gfx1250) — matrix operand order repaired and checked by reading, not compiled, not run on hardware
//
#include <hip/hip_runtime.h>
#include <hip/hip_bf16.h>

#define FEAT 128
#define HID  128

typedef _Float16 h16 __attribute__((ext_vector_type(16)));
typedef float    f8  __attribute__((ext_vector_type(8)));

__device__ __forceinline__ float lrelu(float x) { return x > 0.0f ? x : 0.2f * x; }

// Fused GAT(2-node graph) + FC kernel.
// Each block: 256 threads = 8 waves; each wave handles 16 batch rows.
// GEMMs use v_wmma_f32_16x16x32_f16 (f16 in, f32 accumulate).
__global__ __launch_bounds__(256, 1)
void gat_fused_kernel(const float* __restrict__ x1, const float* __restrict__ x2,
                      const float* __restrict__ Wg, const float* __restrict__ att_s,
                      const float* __restrict__ att_d, const float* __restrict__ bgat,
                      const float* __restrict__ Wfc, const float* __restrict__ bfcp,
                      float* __restrict__ out, int B)
{
    // Weights staged once as f16 (row-major). For H = X @ W^T, the WMMA B-operand
    // column n over a contiguous K-range is exactly a contiguous chunk of row n of W.
    __shared__ _Float16 sWg [HID * FEAT];   // 32 KB
    __shared__ _Float16 sWfc[FEAT * HID];   // 32 KB
    __shared__ _Float16 sM  [8][16 * HID];  // 32 KB, per-wave C->A relayout staging

    const int tid = threadIdx.x;
    for (int i = tid; i < HID * FEAT; i += 256) {
        sWg[i]  = (_Float16)Wg[i];
        sWfc[i] = (_Float16)Wfc[i];
    }
    __syncthreads();

    const int wave = tid >> 5;
    const int lane = tid & 31;
    const int hf   = lane >> 4;   // half-wave: 0 or 1
    const int l16  = lane & 15;

    const long rowBase = (long)blockIdx.x * 128 + (long)wave * 16;

    // Per-lane slices of att vectors / biases: column n = nt*16 + l16
    float as_[8], ad_[8], bg_[8], bf_[8];
#pragma unroll
    for (int nt = 0; nt < 8; ++nt) {
        as_[nt] = att_s[nt * 16 + l16];
        ad_[nt] = att_d[nt * 16 + l16];
        bg_[nt] = bgat [nt * 16 + l16];
        bf_[nt] = bfcp [nt * 16 + l16];
    }

    // A-operand row for this lane (row m = l16 of the wave tile); clamp tail rows
    long arow = rowBase + l16;
    if (arow >= B) arow = (long)B - 1;
    const float* p1 = x1 + arow * FEAT;
    const float* p2 = x2 + arow * FEAT;

    // H1 / H2 accumulators: 8 N-tiles of 16x16 f32 each
    f8 acc1[8] = {};
    f8 acc2[8] = {};

    // ---- GEMM 1+2: H1 = X1 @ Wg^T, H2 = X2 @ Wg^T (K = 128 = 4 steps of 32) ----
#pragma unroll
    for (int kt = 0; kt < 4; ++kt) {
        const int kb = kt * 32;
        // A fragment layout (16-bit A 16x32): lane chunks at K = kb+8*hf and kb+16+8*hf
        h16 a1, a2;
        const float* c0 = p1 + kb + 8 * hf;
        const float* c1 = p1 + kb + 16 + 8 * hf;
        const float* d0 = p2 + kb + 8 * hf;
        const float* d1 = p2 + kb + 16 + 8 * hf;
#pragma unroll
        for (int i = 0; i < 8; ++i) {
            a1[i]     = (_Float16)c0[i];
            a1[8 + i] = (_Float16)c1[i];
            a2[i]     = (_Float16)d0[i];
            a2[8 + i] = (_Float16)d1[i];
        }
#pragma unroll
        for (int nt = 0; nt < 8; ++nt) {
            // B fragment: lane holds column n = nt*16+l16, K chunks at kb+8*hf and kb+16+8*hf
            h16 bw;
            const _Float16* wp = &sWg[(nt * 16 + l16) * FEAT + kb + 8 * hf];
#pragma unroll
            for (int i = 0; i < 16; ++i) bw[i] = wp[i < 8 ? i : i + 8];
            acc1[nt] = __builtin_amdgcn_wmma_f32_16x16x32_f16(false, a1, false, bw,
                                                              (short)0, acc1[nt], false, false);
            acc2[nt] = __builtin_amdgcn_wmma_f32_16x16x32_f16(false, a2, false, bw,
                                                              (short)0, acc2[nt], false, false);
        }
    }

    // ---- attention scalars: s = H . att, reduced across the 16-lane N-group ----
    // C-layout: acc[nt][r] at lane L is element [m = r + 8*hf, n = nt*16 + l16]
    float c1w[8], c2w[8];
#pragma unroll
    for (int r = 0; r < 8; ++r) {
        float p1s = 0.f, p1d = 0.f, p2s = 0.f, p2d = 0.f;
#pragma unroll
        for (int nt = 0; nt < 8; ++nt) {
            p1s += acc1[nt][r] * as_[nt];
            p1d += acc1[nt][r] * ad_[nt];
            p2s += acc2[nt][r] * as_[nt];
            p2d += acc2[nt][r] * ad_[nt];
        }
#pragma unroll
        for (int m = 1; m < 16; m <<= 1) {
            p1s += __shfl_xor(p1s, m, 32);
            p1d += __shfl_xor(p1d, m, 32);
            p2s += __shfl_xor(p2s, m, 32);
            p2d += __shfl_xor(p2d, m, 32);
        }
        // node 0: edges 1->0 (H2) and 0->0 (H1)
        float e10 = lrelu(p2s + p1d);
        float e00 = lrelu(p1s + p1d);
        float mx0 = fmaxf(e10, e00);
        float x10 = __expf(e10 - mx0), x00 = __expf(e00 - mx0);
        float inv0 = 1.0f / (x10 + x00);
        float a00 = x10 * inv0, a01 = x00 * inv0;   // a00 -> H2, a01 -> H1
        // node 1: edges 0->1 (H1) and 1->1 (H2)
        float e01 = lrelu(p1s + p2d);
        float e11 = lrelu(p2s + p2d);
        float mx1 = fmaxf(e01, e11);
        float x01 = __expf(e01 - mx1), x11 = __expf(e11 - mx1);
        float inv1 = 1.0f / (x01 + x11);
        float a10 = x01 * inv1, a11 = x11 * inv1;   // a10 -> H1, a11 -> H2
        c1w[r] = 0.5f * (a01 + a10);  // H1 coefficient
        c2w[r] = 0.5f * (a00 + a11);  // H2 coefficient
    }

    // M = c1*H1 + c2*H2 + b_gat  (in place into acc1)
#pragma unroll
    for (int nt = 0; nt < 8; ++nt)
#pragma unroll
        for (int r = 0; r < 8; ++r)
            acc1[nt][r] = c1w[r] * acc1[nt][r] + c2w[r] * acc2[nt][r] + bg_[nt];

    // ---- relayout M (C-layout) -> A-fragments via per-wave LDS staging ----
    _Float16* ms = sM[wave];
#pragma unroll
    for (int nt = 0; nt < 8; ++nt)
#pragma unroll
        for (int r = 0; r < 8; ++r) {
            const int m = r + 8 * hf;
            ms[m * HID + nt * 16 + l16] = (_Float16)acc1[nt][r];
        }
    // cross-lane LDS hazard within the wave: wait for DS stores before re-reading
    asm volatile("s_wait_dscnt 0" ::: "memory");

    // ---- GEMM 3: OUT = M @ Wfc^T + b_fc ----
    f8 oacc[8] = {};
#pragma unroll
    for (int kt = 0; kt < 4; ++kt) {
        const int kb = kt * 32;
        h16 af;
        const _Float16* q0 = ms + l16 * HID + kb + 8 * hf;
        const _Float16* q1 = ms + l16 * HID + kb + 16 + 8 * hf;
#pragma unroll
        for (int i = 0; i < 8; ++i) {
            af[i]     = q0[i];
            af[8 + i] = q1[i];
        }
#pragma unroll
        for (int nt = 0; nt < 8; ++nt) {
            h16 bw;
            const _Float16* wp = &sWfc[(nt * 16 + l16) * HID + kb + 8 * hf];
#pragma unroll
            for (int i = 0; i < 16; ++i) bw[i] = wp[i < 8 ? i : i + 8];
            oacc[nt] = __builtin_amdgcn_wmma_f32_16x16x32_f16(false, af, false, bw,
                                                              (short)0, oacc[nt], false, false);
        }
    }

    // ---- store fp32 output, guarded for tail rows ----
#pragma unroll
    for (int r = 0; r < 8; ++r) {
        const long m = rowBase + r + 8 * hf;
        if (m < B) {
            float* orow = out + m * FEAT;
#pragma unroll
            for (int nt = 0; nt < 8; ++nt)
                orow[nt * 16 + l16] = oacc[nt][r] + bf_[nt];
        }
    }
}

extern "C" void kernel_launch(void* const* d_in, const int* in_sizes, int n_in,
                              void* d_out, int out_size, void* d_ws, size_t ws_size,
                              hipStream_t stream) {
    const float* x1  = (const float*)d_in[0];  // feature1 [B,128]
    const float* x2  = (const float*)d_in[1];  // feature2 [B,128]
    const float* Wg  = (const float*)d_in[2];  // W_gat [128,128]
    const float* as_ = (const float*)d_in[3];  // att_src [128]
    const float* ad_ = (const float*)d_in[4];  // att_dst [128]
    const float* bg  = (const float*)d_in[5];  // b_gat [128]
    const float* Wfc = (const float*)d_in[6];  // W_fc [128,128]
    const float* bfc = (const float*)d_in[7];  // b_fc [128]
    float* out = (float*)d_out;

    const int B = in_sizes[0] / FEAT;
    const int grid = (B + 127) / 128;  // 128 rows per block (8 waves x 16 rows)
    gat_fused_kernel<<<grid, 256, 0, stream>>>(x1, x2, Wg, as_, ad_, bg, Wfc, bfc, out, B);
}
